// ParallelBlock_63900523430693
// MI455X (gfx1250) — compile-verified, matrix operand order checked by reading, not run on hardware
//
#include <hip/hip_runtime.h>

// ---------------------------------------------------------------------------
// ParallelBlock (shifted-window MSA + channel MSA + fusion + MLP) for gfx1250
// bf16 WMMA (v_wmma_f32_16x16x32_bf16) for all dense GEMMs + window attention.
// ---------------------------------------------------------------------------

#define NTOK 65536      // B*H*W
#define CDIM 256
#define SCALEF 0.17677669529663687f   // 32^-0.5

typedef __attribute__((ext_vector_type(16))) __bf16 v16bf;
typedef __attribute__((ext_vector_type(8)))  float  v8f;

union Frag { unsigned u[8]; v16bf v; };
static_assert(sizeof(Frag) == 32, "frag size");

#define FLAG_GELU 1
#define FLAG_RES  2
#define FLAG_F32  4

// Async global->LDS staging if the toolchain exposes it (ASYNCcnt path).
#if defined(__has_builtin)
#  if __has_builtin(__builtin_amdgcn_global_load_async_to_lds_b128)
#    define HAVE_ASYNC_LDS 1
#  endif
#endif
#ifndef HAVE_ASYNC_LDS
#  define HAVE_ASYNC_LDS 0
#endif

#if HAVE_ASYNC_LDS
// Signature (from hipcc diagnostic): param 1 is int-vector(16B)* in AS1.
typedef int vi4 __attribute__((vector_size(16)));
typedef __attribute__((address_space(1))) vi4 gvi4;
typedef __attribute__((address_space(3))) vi4 lvi4;
#endif

__device__ __forceinline__ unsigned short f2bf(float f) {
  unsigned x = __float_as_uint(f);
  x += 0x7FFFu + ((x >> 16) & 1u);          // round-to-nearest-even
  return (unsigned short)(x >> 16);
}
__device__ __forceinline__ float bf2f(unsigned short h) {
  return __uint_as_float(((unsigned)h) << 16);
}
__device__ __forceinline__ float gelu_f(float v) {
  return 0.5f * v * (1.f + erff(v * 0.70710678118654752f));   // exact erf GELU
}
__device__ __forceinline__ float wave_sum32(float v) {
#pragma unroll
  for (int m = 16; m >= 1; m >>= 1) v += __shfl_xor(v, m, 32);
  return v;
}

// Stage 16 bytes global -> LDS (async if available, else vector load + store)
__device__ __forceinline__ void stage16(const unsigned short* __restrict__ g,
                                        unsigned short* l) {
#if HAVE_ASYNC_LDS
  __builtin_amdgcn_global_load_async_to_lds_b128((gvi4*)g, (lvi4*)l, 0, 0);
#else
  *(uint4*)l = *(const uint4*)g;
#endif
}
__device__ __forceinline__ void stage_fence() {
#if HAVE_ASYNC_LDS
  asm volatile("s_wait_asynccnt 0" ::: "memory");
#endif
}

// ---------------------------------------------------------------------------
// Weight conversion: f32 [K,N] -> bf16 transposed [N,K] so GEMM B-fragments
// are contiguous dword pairs (no ds_load_u16 + pack in the inner loop).
// ---------------------------------------------------------------------------
__global__ void cvt_t_kernel(const float* __restrict__ in,
                             unsigned short* __restrict__ out, int K, int N) {
  int idx = blockIdx.x * 256 + threadIdx.x;
  if (idx < K * N) {
    int n = idx / K, k = idx - n * K;
    out[idx] = f2bf(in[(size_t)k * N + n]);
  }
}

// ---------------------------------------------------------------------------
// Dual LayerNorm: one wave per token. Emits window-path bf16 (rolled +
// window-rearranged row order) and channel-path bf16 (+ch_pos), one pass.
// ---------------------------------------------------------------------------
__global__ __launch_bounds__(256) void ln_dual_kernel(
    const float* __restrict__ x,
    const float* __restrict__ g1, const float* __restrict__ b1,
    const float* __restrict__ gc, const float* __restrict__ bc,
    const float* __restrict__ chpos,
    unsigned short* __restrict__ wln, unsigned short* __restrict__ cln) {
  const int wave = threadIdx.x >> 5, lane = threadIdx.x & 31;
  const int tok = blockIdx.x * 8 + wave;
  const float* row = x + (size_t)tok * CDIM;
  float v[8];
#pragma unroll
  for (int j = 0; j < 8; j++) v[j] = row[lane + j * 32];
  float s = 0.f;
#pragma unroll
  for (int j = 0; j < 8; j++) s += v[j];
  const float mean = wave_sum32(s) * (1.f / 256.f);
  float q = 0.f;
#pragma unroll
  for (int j = 0; j < 8; j++) { float d = v[j] - mean; q += d * d; }
  const float rstd = rsqrtf(wave_sum32(q) * (1.f / 256.f) + 1e-5f);

  // token -> (roll -4, window-major) row index for the window path
  const int b = tok >> 14, hw_ = tok & 16383, h = hw_ >> 7, w = hw_ & 127;
  const int h2 = (h - 4) & 127, w2 = (w - 4) & 127;
  const int wrow = (b * 256 + ((h2 >> 3) * 16 + (w2 >> 3))) * 64 +
                   ((h2 & 7) * 8 + (w2 & 7));
#pragma unroll
  for (int j = 0; j < 8; j++) {
    const int c = lane + j * 32;
    const float n = (v[j] - mean) * rstd;
    wln[(size_t)wrow * CDIM + c] = f2bf(n * g1[c] + b1[c]);
    cln[(size_t)tok * CDIM + c]  = f2bf(n * gc[c] + bc[c] + chpos[c]);
  }
}

// LayerNorm for MLP input (fp32 in, bf16 out, natural order)
__global__ __launch_bounds__(256) void ln2_kernel(
    const float* __restrict__ x, const float* __restrict__ g,
    const float* __restrict__ b, unsigned short* __restrict__ out) {
  const int wave = threadIdx.x >> 5, lane = threadIdx.x & 31;
  const int tok = blockIdx.x * 8 + wave;
  const float* row = x + (size_t)tok * CDIM;
  float v[8];
#pragma unroll
  for (int j = 0; j < 8; j++) v[j] = row[lane + j * 32];
  float s = 0.f;
#pragma unroll
  for (int j = 0; j < 8; j++) s += v[j];
  const float mean = wave_sum32(s) * (1.f / 256.f);
  float q = 0.f;
#pragma unroll
  for (int j = 0; j < 8; j++) { float d = v[j] - mean; q += d * d; }
  const float rstd = rsqrtf(wave_sum32(q) * (1.f / 256.f) + 1e-5f);
#pragma unroll
  for (int j = 0; j < 8; j++) {
    const int c = lane + j * 32;
    out[(size_t)tok * CDIM + c] = f2bf((v[j] - mean) * rstd * g[c] + b[c]);
  }
}

// ---------------------------------------------------------------------------
// Tiled bf16 WMMA GEMM: out = epi(A[M,K] @ Wt[N,K]^T + bias).
// Block tile 64(M) x 128(N), 8 waves (2x4), each wave a 32x32 tile (4 WMMAs
// per k-step from 4 fragments). Both A and B fragments are contiguous dword
// LDS reads (weights pre-transposed). FLAGS compile-time -> branch-free epi.
// ---------------------------------------------------------------------------
template <int FLAGS>
__global__ __launch_bounds__(256, 2) void gemm_bf16_wmma_kernel(
    const unsigned short* __restrict__ A, int lda,     // [M,K] bf16
    const unsigned short* __restrict__ Wt,             // [N,K] bf16 (ld = K)
    const float* __restrict__ bias,
    const float* __restrict__ res, int ldr,
    void* __restrict__ outp, int ldo, int ocol, int K) {
  __shared__ unsigned short As[64 * 40];     // row stride 40 halves (80B)
  __shared__ unsigned short Bs[128 * 40];    // [n][k], row stride 40 halves

  const int t = threadIdx.x;
  const int wave = t >> 5, lane = t & 31;
  const int half = lane >> 4, lr = lane & 15;
  const int bm = blockIdx.y * 64, bn = blockIdx.x * 128;
  const int wm = (wave >> 2) * 32, wn = (wave & 3) * 32;

  v8f acc00, acc01, acc10, acc11;
#pragma unroll
  for (int i = 0; i < 8; i++) {
    acc00[i] = 0.f; acc01[i] = 0.f; acc10[i] = 0.f; acc11[i] = 0.f;
  }

  const int ar = t >> 2, ac = (t & 3) * 8;   // A: 64 rows x 32 halves, 1x b128
  for (int k0 = 0; k0 < K; k0 += 32) {
    stage16(&A[(size_t)(bm + ar) * lda + k0 + ac], &As[ar * 40 + ac]);
#pragma unroll
    for (int s = 0; s < 2; s++) {            // B: 128 rows x 32 halves, 2x b128
      const int idx = t + s * 256;
      const int brr = idx >> 2, bcc = (idx & 3) * 8;
      stage16(&Wt[(size_t)(bn + brr) * K + k0 + bcc], &Bs[brr * 40 + bcc]);
    }
#if !HAVE_ASYNC_LDS
    if (k0 + 32 < K) {                       // prefetch next k-tile
      __builtin_prefetch(&A[(size_t)(bm + ar) * lda + k0 + 32 + ac], 0, 1);
      __builtin_prefetch(&Wt[(size_t)(bn + (t >> 2)) * K + k0 + 32], 0, 1);
    }
#endif
    stage_fence();
    __syncthreads();

    Frag a0, a1, b0, b1;
    const unsigned short* pa0 = &As[(wm + lr) * 40];
    const unsigned short* pa1 = &As[(wm + 16 + lr) * 40];
    const unsigned short* pb0 = &Bs[(wn + lr) * 40];
    const unsigned short* pb1 = &Bs[(wn + 16 + lr) * 40];
#pragma unroll
    for (int i = 0; i < 8; i++) {
      const int kk = ((i < 4) ? (i * 2) : (16 + (i - 4) * 2)) + half * 8;
      a0.u[i] = *(const unsigned*)&pa0[kk];
      a1.u[i] = *(const unsigned*)&pa1[kk];
      b0.u[i] = *(const unsigned*)&pb0[kk];
      b1.u[i] = *(const unsigned*)&pb1[kk];
    }
    acc00 = __builtin_amdgcn_wmma_f32_16x16x32_bf16(false, a0.v, false, b0.v,
                                                    (short)0, acc00, false, false);
    acc01 = __builtin_amdgcn_wmma_f32_16x16x32_bf16(false, a0.v, false, b1.v,
                                                    (short)0, acc01, false, false);
    acc10 = __builtin_amdgcn_wmma_f32_16x16x32_bf16(false, a1.v, false, b0.v,
                                                    (short)0, acc10, false, false);
    acc11 = __builtin_amdgcn_wmma_f32_16x16x32_bf16(false, a1.v, false, b1.v,
                                                    (short)0, acc11, false, false);
    __syncthreads();
  }

  // branch-free epilogue (FLAGS is compile-time)
  const int cg0 = bn + wn + lr, cg1 = cg0 + 16;
  const float bv0 = bias[cg0], bv1 = bias[cg1];
  float* outf = (float*)outp;
  unsigned short* outb = (unsigned short*)outp;
#pragma unroll
  for (int r = 0; r < 8; r++) {
    const int m0 = bm + wm + half * 8 + r;
    const int m1 = m0 + 16;
    float v00 = acc00[r] + bv0, v01 = acc01[r] + bv1;
    float v10 = acc10[r] + bv0, v11 = acc11[r] + bv1;
    if (FLAGS & FLAG_GELU) {
      v00 = gelu_f(v00); v01 = gelu_f(v01);
      v10 = gelu_f(v10); v11 = gelu_f(v11);
    }
    if (FLAGS & FLAG_RES) {
      v00 += res[(size_t)m0 * ldr + cg0];
      v01 += res[(size_t)m0 * ldr + cg1];
      v10 += res[(size_t)m1 * ldr + cg0];
      v11 += res[(size_t)m1 * ldr + cg1];
    }
    if (FLAGS & FLAG_F32) {
      outf[(size_t)m0 * ldo + ocol + cg0] = v00;
      outf[(size_t)m0 * ldo + ocol + cg1] = v01;
      outf[(size_t)m1 * ldo + ocol + cg0] = v10;
      outf[(size_t)m1 * ldo + ocol + cg1] = v11;
    } else {
      outb[(size_t)m0 * ldo + ocol + cg0] = f2bf(v00);
      outb[(size_t)m0 * ldo + ocol + cg1] = f2bf(v01);
      outb[(size_t)m1 * ldo + ocol + cg0] = f2bf(v10);
      outb[(size_t)m1 * ldo + ocol + cg1] = f2bf(v11);
    }
  }
}

// ---------------------------------------------------------------------------
// Shifted-window attention: one wave per (b, window, head).
// sim = q k^T (16 WMMAs) + rel-pos bias + SW mask, row softmax via shfl,
// out = probs @ v (LDS-staged probs, chained WMMAs). Inverse roll at write.
// ---------------------------------------------------------------------------
__global__ __launch_bounds__(256, 2) void wattn_kernel(
    const unsigned short* __restrict__ qkv,     // [B*nw*64, 768] window-ordered
    const float* __restrict__ rp,               // [8,15,15]
    unsigned short* __restrict__ outp) {        // [NTOK, 256] natural order
  __shared__ unsigned short Ps[8][16 * 68];

  const int t = threadIdx.x, wave = t >> 5, lane = t & 31;
  const int half = lane >> 4, lr = lane & 15;
  const int task = blockIdx.x * 8 + wave;
  const int head = task & 7;
  const int wi = (task >> 3) & 255;
  const int b = task >> 11;
  const int row0 = (b * 256 + wi) * 64;
  const int qc = head * 32, kc = 256 + head * 32, vc = 512 + head * 32;
  const int hwi = wi >> 4, wwi = wi & 15;
  const float* rph = rp + head * 225;

  // k^T B-fragments (4 col-tiles); pairs (d,d+1) contiguous in k rows
  Frag kf[4];
#pragma unroll
  for (int j = 0; j < 4; j++) {
    const unsigned short* krow = &qkv[(size_t)(row0 + j * 16 + lr) * 768 + kc];
#pragma unroll
    for (int i = 0; i < 8; i++) {
      const int kk = ((i < 4) ? (i * 2) : (16 + (i - 4) * 2)) + half * 8;
      kf[j].u[i] = *(const unsigned*)&krow[kk];
    }
  }
  // v B-fragments [d-tile][k-step]; K runs over q rows -> ushort gathers
  Frag vf[2][2];
#pragma unroll
  for (int jd = 0; jd < 2; jd++)
#pragma unroll
    for (int ks = 0; ks < 2; ks++)
#pragma unroll
      for (int i = 0; i < 8; i++) {
        const int q0 = ((i < 4) ? (i * 2) : (16 + (i - 4) * 2)) + half * 8 + ks * 32;
        const unsigned lo = qkv[(size_t)(row0 + q0) * 768 + vc + jd * 16 + lr];
        const unsigned hi = qkv[(size_t)(row0 + q0 + 1) * 768 + vc + jd * 16 + lr];
        vf[jd][ks].u[i] = lo | (hi << 16);
      }

  for (int ib = 0; ib < 4; ib++) {              // 16-row blocks of the window
    Frag aq;
    const unsigned short* qrow = &qkv[(size_t)(row0 + ib * 16 + lr) * 768 + qc];
#pragma unroll
    for (int i = 0; i < 8; i++) {
      const int kk = ((i < 4) ? (i * 2) : (16 + (i - 4) * 2)) + half * 8;
      aq.u[i] = *(const unsigned*)&qrow[kk];
    }
    v8f sim[4];
#pragma unroll
    for (int j = 0; j < 4; j++) {
      v8f z;
#pragma unroll
      for (int i = 0; i < 8; i++) z[i] = 0.f;
      sim[j] = __builtin_amdgcn_wmma_f32_16x16x32_bf16(false, aq.v, false,
                                                       kf[j].v, (short)0, z,
                                                       false, false);
    }
    // bias + shifted-window mask + row softmax (rows live on 16-lane halves)
#pragma unroll
    for (int r = 0; r < 8; r++) {
      const int p = ib * 16 + half * 8 + r;
      const int pi = p >> 3, pc = p & 7;
      float sj[4];
      float mx = -3.0e38f;
#pragma unroll
      for (int j = 0; j < 4; j++) {
        const int n = j * 16 + lr;
        const int qi = n >> 3, qcc = n & 7;
        float s = sim[j][r] * SCALEF + rph[(pi - qi + 7) * 15 + (pc - qcc + 7)];
        const bool msk = (hwi == 15 && ((pi < 4) != (qi < 4))) ||
                         (wwi == 15 && ((pc < 4) != (qcc < 4)));
        s = msk ? -3.0e38f : s;
        sj[j] = s;
        mx = fmaxf(mx, s);
      }
#pragma unroll
      for (int m = 1; m < 16; m <<= 1) mx = fmaxf(mx, __shfl_xor(mx, m, 32));
      float sum = 0.f;
#pragma unroll
      for (int j = 0; j < 4; j++) { sj[j] = __expf(sj[j] - mx); sum += sj[j]; }
#pragma unroll
      for (int m = 1; m < 16; m <<= 1) sum += __shfl_xor(sum, m, 32);
      const float inv = 1.f / sum;
      const int rowl = half * 8 + r;
#pragma unroll
      for (int j = 0; j < 4; j++)
        Ps[wave][rowl * 68 + j * 16 + lr] = f2bf(sj[j] * inv);
    }
    __syncthreads();

    Frag ap0, ap1;
#pragma unroll
    for (int i = 0; i < 8; i++) {
      const int kk = ((i < 4) ? (i * 2) : (16 + (i - 4) * 2)) + half * 8;
      ap0.u[i] = *(const unsigned*)&Ps[wave][lr * 68 + kk];
      ap1.u[i] = *(const unsigned*)&Ps[wave][lr * 68 + kk + 32];
    }
#pragma unroll
    for (int jd = 0; jd < 2; jd++) {
      v8f oc;
#pragma unroll
      for (int i = 0; i < 8; i++) oc[i] = 0.f;
      oc = __builtin_amdgcn_wmma_f32_16x16x32_bf16(false, ap0.v, false,
                                                   vf[jd][0].v, (short)0, oc,
                                                   false, false);
      oc = __builtin_amdgcn_wmma_f32_16x16x32_bf16(false, ap1.v, false,
                                                   vf[jd][1].v, (short)0, oc,
                                                   false, false);
#pragma unroll
      for (int r = 0; r < 8; r++) {
        const int p = ib * 16 + half * 8 + r;
        const int h2 = hwi * 8 + (p >> 3), w2 = wwi * 8 + (p & 7);
        const int h = (h2 + 4) & 127, w = (w2 + 4) & 127;
        const size_t tok = (size_t)b * 16384 + h * 128 + w;
        outp[tok * CDIM + head * 32 + jd * 16 + lr] = f2bf(oc[r]);
      }
    }
    __syncthreads();
  }
}

// ---------------------------------------------------------------------------
// Channel attention logits + softmax: 32x32 per (b,head), K-reduction 16384.
// Tiny FLOP fraction (~1.7%) -> LDS-tiled VALU dot products are the right cost.
// ---------------------------------------------------------------------------
__global__ __launch_bounds__(256) void cattn_logits_kernel(
    const unsigned short* __restrict__ qkv,     // [NTOK, 768] natural order
    unsigned short* __restrict__ probs) {       // [B*8, 32, 32] bf16
  __shared__ unsigned short Qs[64 * 32];
  __shared__ unsigned short Ks[64 * 32];
  __shared__ float Attn[32 * 32];
  const int t = threadIdx.x;
  const int bh = blockIdx.x;                    // b*8 + head
  const int b = bh >> 3, head = bh & 7;
  const size_t base = (size_t)b * 16384;
  const int r = t >> 2, ch = (t & 3) * 8;
  const int d = t >> 3, e0 = (t & 7) * 4;
  float acc[4] = {0.f, 0.f, 0.f, 0.f};

  for (int l0 = 0; l0 < 16384; l0 += 64) {
    *(uint4*)&Qs[r * 32 + ch] =
        *(const uint4*)&qkv[(base + l0 + r) * 768 + head * 32 + ch];
    *(uint4*)&Ks[r * 32 + ch] =
        *(const uint4*)&qkv[(base + l0 + r) * 768 + 256 + head * 32 + ch];
    __syncthreads();
    for (int dl = 0; dl < 64; dl++) {
      const float qv = bf2f(Qs[dl * 32 + d]);
#pragma unroll
      for (int j = 0; j < 4; j++) acc[j] += qv * bf2f(Ks[dl * 32 + e0 + j]);
    }
    __syncthreads();
  }
#pragma unroll
  for (int j = 0; j < 4; j++) Attn[d * 32 + e0 + j] = acc[j] * SCALEF;
  __syncthreads();
  if (t < 32) {
    float mx = -3.0e38f;
    for (int e = 0; e < 32; e++) mx = fmaxf(mx, Attn[t * 32 + e]);
    float sum = 0.f, ex[32];
    for (int e = 0; e < 32; e++) { ex[e] = __expf(Attn[t * 32 + e] - mx); sum += ex[e]; }
    const float inv = 1.f / sum;
    for (int e = 0; e < 32; e++)
      probs[((size_t)bh * 32 + t) * 32 + e] = f2bf(ex[e] * inv);
  }
}

// o[l, head*32+d] = sum_e probs[d,e] * v[e,l]; transpose to [N,C] at the write
__global__ __launch_bounds__(256) void cattn_out_kernel(
    const unsigned short* __restrict__ qkv,
    const unsigned short* __restrict__ probs,
    unsigned short* __restrict__ outp) {
  const size_t tok = blockIdx.x;
  const int t = threadIdx.x;
  const int head = t >> 5, d = t & 31;
  const int b = (int)(tok >> 14);
  const unsigned short* vrow = &qkv[tok * 768 + 512 + head * 32];
  const unsigned short* pr = &probs[(size_t)((b * 8 + head) * 32 + d) * 32];
  float s = 0.f;
#pragma unroll
  for (int e = 0; e < 32; e++) s += bf2f(pr[e]) * bf2f(vrow[e]);
  outp[tok * CDIM + t] = f2bf(s);
}

// ---------------------------------------------------------------------------
extern "C" void kernel_launch(void* const* d_in, const int* in_sizes, int n_in,
                              void* d_out, int out_size, void* d_ws, size_t ws_size,
                              hipStream_t stream) {
  (void)in_sizes; (void)n_in; (void)out_size; (void)ws_size;
  const float* x      = (const float*)d_in[0];
  const float* ln1_g  = (const float*)d_in[1];
  const float* ln1_b  = (const float*)d_in[2];
  const float* wq_w   = (const float*)d_in[3];
  const float* wq_b   = (const float*)d_in[4];
  const float* rp     = (const float*)d_in[5];
  const float* wo_w   = (const float*)d_in[6];
  const float* wo_b   = (const float*)d_in[7];
  const float* lnc_g  = (const float*)d_in[8];
  const float* lnc_b  = (const float*)d_in[9];
  const float* ch_pos = (const float*)d_in[10];
  const float* cq_w   = (const float*)d_in[11];
  const float* cq_b   = (const float*)d_in[12];
  const float* cp_w   = (const float*)d_in[13];
  const float* cp_b   = (const float*)d_in[14];
  const float* fus_w  = (const float*)d_in[15];
  const float* fus_b  = (const float*)d_in[16];
  const float* ln2_g  = (const float*)d_in[17];
  const float* ln2_b  = (const float*)d_in[18];
  const float* m_w1   = (const float*)d_in[19];
  const float* m_b1   = (const float*)d_in[20];
  const float* m_w2   = (const float*)d_in[21];
  const float* m_b2   = (const float*)d_in[22];

  char* base = (char*)d_ws;
  size_t off = 0;
  auto take = [&](size_t bytes) -> void* {
    void* p = base + off;
    off = (off + bytes + 255) & ~(size_t)255;
    return p;
  };
  // transposed bf16 weights [N,K]
  unsigned short* wq_bf  = (unsigned short*)take((size_t)768 * 256 * 2);
  unsigned short* wo_bf  = (unsigned short*)take((size_t)256 * 256 * 2);
  unsigned short* cq_bf  = (unsigned short*)take((size_t)768 * 256 * 2);
  unsigned short* cp_bf  = (unsigned short*)take((size_t)256 * 256 * 2);
  unsigned short* fus_bf = (unsigned short*)take((size_t)256 * 512 * 2);
  unsigned short* mw1_bf = (unsigned short*)take((size_t)512 * 256 * 2);
  unsigned short* mw2_bf = (unsigned short*)take((size_t)256 * 512 * 2);
  unsigned short* wln  = (unsigned short*)take((size_t)NTOK * 256 * 2);  // reused as yln
  unsigned short* cln  = (unsigned short*)take((size_t)NTOK * 256 * 2);
  unsigned short* qkvb = (unsigned short*)take((size_t)NTOK * 768 * 2);  // shared W then C path
  unsigned short* atb  = (unsigned short*)take((size_t)NTOK * 256 * 2);  // attn out, shared
  unsigned short* catb = (unsigned short*)take((size_t)NTOK * 512 * 2);  // concat, reused as h1
  float*          xf   = (float*)take((size_t)NTOK * 256 * 4);
  unsigned short* cprob = (unsigned short*)take((size_t)32 * 32 * 32 * 2);

  auto cvt_t = [&](const float* src, unsigned short* dst, int K, int N) {
    cvt_t_kernel<<<dim3((K * N + 255) / 256), dim3(256), 0, stream>>>(src, dst, K, N);
  };
  cvt_t(wq_w,  wq_bf,  256, 768);
  cvt_t(wo_w,  wo_bf,  256, 256);
  cvt_t(cq_w,  cq_bf,  256, 768);
  cvt_t(cp_w,  cp_bf,  256, 256);
  cvt_t(fus_w, fus_bf, 512, 256);
  cvt_t(m_w1,  mw1_bf, 256, 512);
  cvt_t(m_w2,  mw2_bf, 512, 256);

  const dim3 blk(256);
  auto grid_for = [&](int N_) { return dim3(N_ / 128, NTOK / 64); };

  // 1. dual LN (window-reordered + channel w/ ch_pos)
  ln_dual_kernel<<<dim3(NTOK / 8), blk, 0, stream>>>(
      x, ln1_g, ln1_b, lnc_g, lnc_b, ch_pos, wln, cln);

  // 2. window path: qkv, attention, out-proj (+x) -> cat[:, :256]
  gemm_bf16_wmma_kernel<0><<<grid_for(768), blk, 0, stream>>>(
      wln, 256, wq_bf, wq_b, nullptr, 0, qkvb, 768, 0, 256);
  wattn_kernel<<<dim3(1024), blk, 0, stream>>>(qkvb, rp, atb);
  gemm_bf16_wmma_kernel<FLAG_RES><<<grid_for(256), blk, 0, stream>>>(
      atb, 256, wo_bf, wo_b, x, 256, catb, 512, 0, 256);

  // 3. channel path: qkv, channel attention, proj (+x) -> cat[:, 256:]
  gemm_bf16_wmma_kernel<0><<<grid_for(768), blk, 0, stream>>>(
      cln, 256, cq_bf, cq_b, nullptr, 0, qkvb, 768, 0, 256);
  cattn_logits_kernel<<<dim3(32), blk, 0, stream>>>(qkvb, cprob);
  cattn_out_kernel<<<dim3(NTOK), blk, 0, stream>>>(qkvb, cprob, atb);
  gemm_bf16_wmma_kernel<FLAG_RES><<<grid_for(256), blk, 0, stream>>>(
      atb, 256, cp_bf, cp_b, x, 256, catb, 512, 256, 256);

  // 4. fusion: xf = x + gelu(cat @ fus_w + fus_b)   (f32)
  gemm_bf16_wmma_kernel<FLAG_GELU | FLAG_RES | FLAG_F32>
      <<<grid_for(256), blk, 0, stream>>>(
      catb, 512, fus_bf, fus_b, x, 256, xf, 256, 0, 512);

  // 5. MLP: out = xf + gelu(LN(xf) @ m_w1 + b1) @ m_w2 + b2
  ln2_kernel<<<dim3(NTOK / 8), blk, 0, stream>>>(xf, ln2_g, ln2_b, wln);
  gemm_bf16_wmma_kernel<FLAG_GELU><<<grid_for(512), blk, 0, stream>>>(
      wln, 256, mw1_bf, m_b1, nullptr, 0, catb, 512, 0, 256);
  gemm_bf16_wmma_kernel<FLAG_RES | FLAG_F32><<<grid_for(256), blk, 0, stream>>>(
      catb, 512, mw2_bf, m_b2, xf, 256, d_out, 256, 0, 512);
}
